// Encoder_39213051412927
// MI455X (gfx1250) — matrix operand order repaired and checked by reading, not compiled, not run on hardware
//
#include <hip/hip_runtime.h>
#include <hip/hip_bf16.h>

#define ASYNC_COPY 1   // use global_load_async_to_lds_b128 for LDS staging

// ---------------------------------------------------------------------------
// Types for CDNA5 WMMA (wave32): 16x16x32 bf16, f32 accumulate.
// ---------------------------------------------------------------------------
typedef __attribute__((ext_vector_type(16))) __bf16 v16bf;
typedef __attribute__((ext_vector_type(8)))  float  v8f;

union BF16x16 {
    unsigned int u[8];
    v16bf        v;
};

__device__ __forceinline__ unsigned int bf16_rne(float f) {
    unsigned int u = __float_as_uint(f);
    u += 0x7FFFu + ((u >> 16) & 1u);   // round-to-nearest-even
    return u >> 16;
}
__device__ __forceinline__ unsigned int pack_bf16(float lo, float hi) {
    return bf16_rne(lo) | (bf16_rne(hi) << 16);
}

// Load four packed 32x16 bf16 B-tiles from a pointer (global or LDS).
__device__ __forceinline__ void load4(const unsigned int* __restrict__ p,
                                      BF16x16* t) {
#pragma unroll
    for (int n = 0; n < 4; ++n) {
        const uint4* q = (const uint4*)(p + n * 256);
        uint4 x0 = q[0], x1 = q[1];
        t[n].u[0] = x0.x; t[n].u[1] = x0.y; t[n].u[2] = x0.z; t[n].u[3] = x0.w;
        t[n].u[4] = x1.x; t[n].u[5] = x1.y; t[n].u[6] = x1.z; t[n].u[7] = x1.w;
    }
}

// A-fragment = diag(scale) * row fragments, via v_pk_mul_bf16 (co-execs w/ XDL)
__device__ __forceinline__ BF16x16 scaleA(const unsigned int* ab, unsigned int sc) {
    BF16x16 a;
#pragma unroll
    for (int v = 0; v < 8; ++v) {
        unsigned int m;
        asm("v_pk_mul_bf16 %0, %1, %2" : "=v"(m) : "v"(ab[v]), "v"(sc));
        a.u[v] = m;
    }
    return a;
}

#define WMMA4(AV, BT)                                                          \
    c0 = __builtin_amdgcn_wmma_f32_16x16x32_bf16(false, (AV), false, (BT)[0].v, \
                                                 (short)0, c0, false, false);   \
    c1 = __builtin_amdgcn_wmma_f32_16x16x32_bf16(false, (AV), false, (BT)[1].v, \
                                                 (short)0, c1, false, false);   \
    c2 = __builtin_amdgcn_wmma_f32_16x16x32_bf16(false, (AV), false, (BT)[2].v, \
                                                 (short)0, c2, false, false);   \
    c3 = __builtin_amdgcn_wmma_f32_16x16x32_bf16(false, (AV), false, (BT)[3].v, \
                                                 (short)0, c3, false, false);

#define CDIV(a, b) (((a) + (b) - 1) / (b))

// ---------------------------------------------------------------------------
// Pack a row-major f32 [K, N] B-matrix into bf16 WMMA B-register layout:
//   out[((kb*(N/16)+nb)*32 + lane)*8 + v] packs K = kb*32 + (v>>2)*16 + (lane>>4)*8 + 2(v&3), +1
//   at column n = nb*16 + (lane&15).
// ---------------------------------------------------------------------------
__global__ void k_pack_b(const float* __restrict__ W,
                         unsigned int* __restrict__ outp,
                         int K, int Nn, int total) {
    int tid = blockIdx.x * blockDim.x + threadIdx.x;
    if (tid >= total) return;
    int nb_cnt = Nn >> 4;
    int v    = tid & 7;
    int lane = (tid >> 3) & 31;
    int kn   = tid >> 8;               // kb*nb_cnt + nb
    int nb   = kn % nb_cnt;
    int kb   = kn / nb_cnt;
    int n = nb * 16 + (lane & 15);
    int k = kb * 32 + (v >> 2) * 16 + (lane >> 4) * 8 + 2 * (v & 3);
    outp[tid] = pack_bf16(W[(size_t)k * Nn + n], W[(size_t)(k + 1) * Nn + n]);
}

// ---------------------------------------------------------------------------
// lin0:  out[N,64] = relu(x[N,768] @ W0 + b0), bf16 WMMA, one wave per 16 rows.
// (memory-bound on x: 61 MB @ 23.3 TB/s dominates; scheduling is secondary)
// ---------------------------------------------------------------------------
__global__ void k_lin0(const float* __restrict__ x,
                       const unsigned int* __restrict__ Wpk,
                       const float* __restrict__ b0,
                       float* __restrict__ out, int Nn) {
    int wid  = (int)((blockIdx.x * (size_t)blockDim.x + threadIdx.x) >> 5);
    int lane = (int)(threadIdx.x & 31);
    int ntiles = (Nn + 15) >> 4;
    if (wid >= ntiles) return;                      // wave-uniform exit
    int hh = lane >> 4, mr = lane & 15;
    int row = wid * 16 + mr;
    const float* xr = x + (size_t)min(row, Nn - 1) * 768;
    v8f c0 = {}, c1 = {}, c2 = {}, c3 = {};
    const unsigned int* tb = Wpk + (size_t)lane * 8;
    for (int kc = 0; kc < 24; ++kc) {
        const float4* p0 = (const float4*)(xr + kc * 32 + hh * 8);
        const float4* p1 = (const float4*)(xr + kc * 32 + 16 + hh * 8);
        float4 q0 = p0[0], q1 = p0[1], q2 = p1[0], q3 = p1[1];
        BF16x16 bt[4];
        load4(tb, bt);
        BF16x16 a;
        a.u[0] = pack_bf16(q0.x, q0.y); a.u[1] = pack_bf16(q0.z, q0.w);
        a.u[2] = pack_bf16(q1.x, q1.y); a.u[3] = pack_bf16(q1.z, q1.w);
        a.u[4] = pack_bf16(q2.x, q2.y); a.u[5] = pack_bf16(q2.z, q2.w);
        a.u[6] = pack_bf16(q3.x, q3.y); a.u[7] = pack_bf16(q3.z, q3.w);
        WMMA4(a.v, bt);
        tb += 1024;
    }
    float bb0 = b0[mr], bb1 = b0[16 + mr], bb2 = b0[32 + mr], bb3 = b0[48 + mr];
#pragma unroll
    for (int r = 0; r < 8; ++r) {
        int orow = wid * 16 + r + 8 * hh;           // C: M = r + 8*(lane>>4)
        if (orow < Nn) {
            float* po = out + (size_t)orow * 64 + mr;
            po[0]  = fmaxf(c0[r] + bb0, 0.f);
            po[16] = fmaxf(c1[r] + bb1, 0.f);
            po[32] = fmaxf(c2[r] + bb2, 0.f);
            po[48] = fmaxf(c3[r] + bb3, 0.f);
        }
    }
}

// h = out; out_bf16 = bf16(out)
__global__ void k_seed(const float* __restrict__ src, float* __restrict__ h,
                       unsigned short* __restrict__ outb, int count) {
    int tid = blockIdx.x * blockDim.x + threadIdx.x;
    if (tid >= count) return;
    float v = src[tid];
    h[tid] = v;
    outb[tid] = (unsigned short)bf16_rne(v);
}

// ---------------------------------------------------------------------------
// edge MLP layer 1: h1b[E,64] = bf16(relu(edge_attr[E,85] @ en1_w + en1_b))
// ---------------------------------------------------------------------------
__global__ void k_edge_h1(const float* __restrict__ ea,
                          const float* __restrict__ W,   // [85][64]
                          const float* __restrict__ b,
                          unsigned short* __restrict__ h1b, int E) {
    __shared__ float ws[85 * 64];
    __shared__ float bs[64];
    for (int i = threadIdx.x; i < 85 * 64; i += blockDim.x) ws[i] = W[i];
    if (threadIdx.x < 64) bs[threadIdx.x] = b[threadIdx.x];
    __syncthreads();
    int e = blockIdx.x * blockDim.x + threadIdx.x;
    if (e >= E) return;
    const float* ar = ea + (size_t)e * 85;
    float acc[64];
#pragma unroll
    for (int o = 0; o < 64; ++o) acc[o] = bs[o];
    for (int k = 0; k < 85; ++k) {
        float a = ar[k];
#pragma unroll
        for (int o = 0; o < 64; ++o) acc[o] += a * ws[k * 64 + o];
    }
    unsigned int* op = (unsigned int*)(h1b + (size_t)e * 64);
#pragma unroll
    for (int j = 0; j < 32; ++j)
        op[j] = pack_bf16(fmaxf(acc[2 * j], 0.f), fmaxf(acc[2 * j + 1], 0.f));
}

__global__ void k_zero(float* __restrict__ p, int count) {
    int tid = blockIdx.x * blockDim.x + threadIdx.x;
    if (tid < count) p[tid] = 0.f;
}

__global__ void k_count(const int* __restrict__ dst, float* __restrict__ cnt, int E) {
    int e = blockIdx.x * blockDim.x + threadIdx.x;
    if (e < E) atomicAdd(&cnt[dst[e]], 1.0f);
}

// ---------------------------------------------------------------------------
// Fused NNConv message kernel:  msg[e,:] = out[src[e],:] @ (h1_e @ en2_w + en2_b)
// as GEMM  P[E,4096] @ T[4096,64], P rows rank-1-structured & built in-register:
//   K'-chunk (32 wide) = fixed k, 32-run of i -> A = diag(h1[e,k]) * out_src frags.
// Block = 8 waves = 128 edges share T through LDS:
//   130 packed chunks (128 en2_w + 2 en2_b) staged as 16 x 8-chunk (32 KB)
//   stages + 2-chunk bias tail, double-buffered in LDS. Stage copies are issued
//   with global_load_async_to_lds_b128 (ASYNCcnt) and overlap the current
//   stage's WMMA work; s_wait_asynccnt 0 + barrier close each stage.
// Global B traffic: 532 KB per 128 edges (8x less than per-wave streaming).
// Scatter-adds msg into agg[dst] with global f32 atomics.
// ---------------------------------------------------------------------------
__global__ void __launch_bounds__(256)
k_msg(const unsigned short* __restrict__ outb,  // [N][64] bf16
      const unsigned short* __restrict__ h1b,   // [E][64] bf16
      const int* __restrict__ src,
      const int* __restrict__ dst,
      const unsigned int* __restrict__ Tpk,     // 130 packed chunks x 1024 u32
      float* __restrict__ agg, int E) {
    __shared__ __align__(16) unsigned int sbuf[2][8192];   // 2 x 32 KB
    const int tid  = threadIdx.x;
    const int lane = tid & 31;
    const int wv   = tid >> 5;                  // wave in block: 0..7
    const int e0   = blockIdx.x * 128 + wv * 16;
    const int hh = lane >> 4, mr = lane & 15;
    const int e  = min(e0 + mr, E - 1);
    const int s  = src[e];

    // out_src row fragments in A-layout for both K'-chunk parities.
    const uint4* orow = (const uint4*)(outb + (size_t)s * 64);
    uint4 r0 = orow[hh], r1 = orow[2 + hh], r2 = orow[4 + hh], r3 = orow[6 + hh];
    unsigned int a0[8] = {r0.x, r0.y, r0.z, r0.w, r1.x, r1.y, r1.z, r1.w}; // i in [0,32)
    unsigned int a1[8] = {r2.x, r2.y, r2.z, r2.w, r3.x, r3.y, r3.z, r3.w}; // i in [32,64)
    const unsigned int* __restrict__ h1row = (const unsigned int*)(h1b + (size_t)e * 64);

    v8f c0 = {}, c1 = {}, c2 = {}, c3 = {};

    // stage copy: stage sg = chunks [sg*8, sg*8+8) (tail stage 16: 2 chunks)
    auto issue_copy = [&](int sg, int bufIdx) {
        int nvec = (sg < 16) ? 8 : 2;           // sweeps of 256 x uint4
        unsigned gbase = (unsigned)sg * 8192u;  // u32 offset into Tpk
#pragma unroll
        for (int j = 0; j < 8; ++j) {
            if (j < nvec) {
                unsigned idx = (unsigned)(j * 256 + tid) * 4u;      // u32 index
#if ASYNC_COPY
                unsigned ldsoff = (unsigned)(size_t)(&sbuf[bufIdx][idx]);
                unsigned goff   = (gbase + idx) * 4u;               // byte offset
                asm volatile("global_load_async_to_lds_b128 %0, %1, %2"
                             :: "v"(ldsoff), "v"(goff), "s"(Tpk) : "memory");
#else
                uint4 q = *(const uint4*)(Tpk + gbase + idx);
                *(uint4*)(&sbuf[bufIdx][idx]) = q;
#endif
            }
        }
    };

    issue_copy(0, 0);
#if ASYNC_COPY
    asm volatile("s_wait_asynccnt 0x0" ::: "memory");
#endif
    __syncthreads();

    for (int st = 0; st < 17; ++st) {
        if (st < 16) issue_copy(st + 1, (st + 1) & 1);  // overlap with compute

        const unsigned int* sb = &sbuf[st & 1][0];
        if (st < 16) {
            unsigned pA = h1row[st * 2];        // bf16 pair: h1[4st], h1[4st+1]
            unsigned pB = h1row[st * 2 + 1];    // bf16 pair: h1[4st+2], h1[4st+3]
#pragma unroll
            for (int c = 0; c < 8; ++c) {       // global chunk g = st*8 + c
                unsigned pr = (c < 4) ? pA : pB;
                unsigned sc = ((c >> 1) & 1) ? ((pr & 0xFFFF0000u) | (pr >> 16))
                                             : ((pr << 16) | (pr & 0xFFFFu));
                const unsigned int* ab = (c & 1) ? a1 : a0;
                BF16x16 bt[4];
                load4(sb + c * 1024 + lane * 8, bt);    // 8 x ds_load_b128
                BF16x16 a = scaleA(ab, sc);
                WMMA4(a.v, bt);
            }
        } else {
            // bias tail: out_src @ reshape(en2_b,[64,64])
#pragma unroll
            for (int c = 0; c < 2; ++c) {
                const unsigned int* ab = (c & 1) ? a1 : a0;
                BF16x16 bt[4];
                load4(sb + c * 1024 + lane * 8, bt);
                BF16x16 a;
#pragma unroll
                for (int v = 0; v < 8; ++v) a.u[v] = ab[v];
                WMMA4(a.v, bt);
            }
        }

        if (st < 16) {
#if ASYNC_COPY
            asm volatile("s_wait_asynccnt 0x0" ::: "memory");
#endif
            __syncthreads();
        }
    }

    // scatter-add: C lane layout row M = r + 8*hh, col = n*16 + mr
#pragma unroll
    for (int r = 0; r < 8; ++r) {
        int erow = e0 + r + 8 * hh;
        if (erow < E) {
            int d = dst[erow];
            float* base = agg + (size_t)d * 64 + mr;
            atomicAdd(base + 0,  c0[r]);
            atomicAdd(base + 16, c1[r]);
            atomicAdd(base + 32, c2[r]);
            atomicAdd(base + 48, c3[r]);
        }
    }
}

// m = relu(agg / max(cnt,1) + conv_b)   (in place on agg)
__global__ void k_m(float* __restrict__ agg, const float* __restrict__ cnt,
                    const float* __restrict__ convb, int Nn) {
    int tid = blockIdx.x * blockDim.x + threadIdx.x;
    if (tid >= Nn * 64) return;
    float den = fmaxf(cnt[tid >> 6], 1.0f);
    float v = agg[tid] / den + convb[tid & 63];
    agg[tid] = fmaxf(v, 0.f);
}

// hnew = relu(h @ linh_w + linh_b)
__global__ void k_linh(const float* __restrict__ h, const float* __restrict__ W,
                       const float* __restrict__ b, float* __restrict__ hnew, int Nn) {
    __shared__ float ws[64 * 64];
    __shared__ float bs[64];
    for (int i = threadIdx.x; i < 64 * 64; i += blockDim.x) ws[i] = W[i];
    if (threadIdx.x < 64) bs[threadIdx.x] = b[threadIdx.x];
    __syncthreads();
    int tid = blockIdx.x * blockDim.x + threadIdx.x;
    if (tid >= Nn * 64) return;
    int n = tid >> 6, o = tid & 63;
    const float* hr = h + (size_t)n * 64;
    float acc = bs[o];
#pragma unroll
    for (int k = 0; k < 64; ++k) acc += hr[k] * ws[k * 64 + o];
    hnew[tid] = fmaxf(acc, 0.f);
}

// out = relu([hnew, m] @ linhm_w + linhm_b) + h_old ;  out_bf16 = bf16(out)
__global__ void k_linhm(const float* __restrict__ hnew, const float* __restrict__ m,
                        const float* __restrict__ hold, const float* __restrict__ W,
                        const float* __restrict__ b, float* __restrict__ out,
                        unsigned short* __restrict__ outb, int Nn) {
    __shared__ float ws[128 * 64];
    __shared__ float bs[64];
    for (int i = threadIdx.x; i < 128 * 64; i += blockDim.x) ws[i] = W[i];
    if (threadIdx.x < 64) bs[threadIdx.x] = b[threadIdx.x];
    __syncthreads();
    int tid = blockIdx.x * blockDim.x + threadIdx.x;
    if (tid >= Nn * 64) return;
    int n = tid >> 6, o = tid & 63;
    const float* hr = hnew + (size_t)n * 64;
    const float* mr = m + (size_t)n * 64;
    float acc = bs[o];
#pragma unroll
    for (int k = 0; k < 64; ++k) acc += hr[k] * ws[k * 64 + o];
#pragma unroll
    for (int k = 0; k < 64; ++k) acc += mr[k] * ws[(64 + k) * 64 + o];
    float v = fmaxf(acc, 0.f) + hold[tid];
    out[tid] = v;
    outb[tid] = (unsigned short)bf16_rne(v);
}

__global__ void k_final(const float* __restrict__ src, float* __restrict__ dstp, int count) {
    int tid = blockIdx.x * blockDim.x + threadIdx.x;
    if (tid < count) dstp[tid] = fmaxf(src[tid], 0.f);
}

// ---------------------------------------------------------------------------
extern "C" void kernel_launch(void* const* d_in, const int* in_sizes, int n_in,
                              void* d_out, int out_size, void* d_ws, size_t ws_size,
                              hipStream_t stream) {
    const float* x        = (const float*)d_in[0];
    const int*   ei       = (const int*)d_in[1];
    const float* edge_attr= (const float*)d_in[2];
    const float* lin0_w   = (const float*)d_in[3];
    const float* lin0_b   = (const float*)d_in[4];
    const float* linh_w   = (const float*)d_in[5];
    const float* linh_b   = (const float*)d_in[6];
    const float* linhm_w  = (const float*)d_in[7];
    const float* linhm_b  = (const float*)d_in[8];
    const float* en1_w    = (const float*)d_in[9];
    const float* en1_b    = (const float*)d_in[10];
    const float* en2_w    = (const float*)d_in[11];
    const float* en2_b    = (const float*)d_in[12];
    const float* conv_b   = (const float*)d_in[13];

    const int N = in_sizes[0] / 768;
    const int E = in_sizes[1] / 2;
    const int* src = ei;
    const int* dst = ei + E;

    // workspace carve-out
    char* w = (char*)d_ws;
    auto alloc = [&](size_t bytes) -> void* {
        void* p = (void*)w;
        w += (bytes + 255) & ~(size_t)255;
        return p;
    };
    float* outf = (float*)alloc((size_t)N * 64 * 4);
    float* h    = (float*)alloc((size_t)N * 64 * 4);
    float* hnew = (float*)alloc((size_t)N * 64 * 4);
    float* agg  = (float*)alloc((size_t)N * 64 * 4);
    float* cnt  = (float*)alloc((size_t)N * 4);
    unsigned short* outb = (unsigned short*)alloc((size_t)N * 64 * 2);
    unsigned short* h1b  = (unsigned short*)alloc((size_t)E * 64 * 2);
    unsigned int* Wpk = (unsigned int*)alloc((size_t)(768 / 32) * 4 * 256 * 4);
    // T (en2_w as [4096,64], 128 chunks) + en2_b (as [64,64], 2 chunks): contiguous
    unsigned int* Tpk = (unsigned int*)alloc((size_t)130 * 1024 * 4);
    unsigned int* Bpk = Tpk + (size_t)128 * 1024;

    const int B = 256;

    // pack weights to bf16 WMMA B-layout
    {
        int t0 = (768 / 32) * 4 * 256;
        k_pack_b<<<CDIV(t0, B), B, 0, stream>>>(lin0_w, Wpk, 768, 64, t0);
        int t1 = (4096 / 32) * 4 * 256;
        k_pack_b<<<CDIV(t1, B), B, 0, stream>>>(en2_w, Tpk, 4096, 64, t1);  // T = en2_w as [4096,64]
        int t2 = (64 / 32) * 4 * 256;
        k_pack_b<<<CDIV(t2, B), B, 0, stream>>>(en2_b, Bpk, 64, 64, t2);    // en2_b as [64,64]
    }

    // in-degree counts
    k_zero<<<CDIV(N, B), B, 0, stream>>>(cnt, N);
    k_count<<<CDIV(E, B), B, 0, stream>>>(dst, cnt, E);

    // lin0 (WMMA) + seed h / out_bf16
    {
        int waves = CDIV(N, 16);
        k_lin0<<<CDIV(waves * 32, B), B, 0, stream>>>(x, Wpk, lin0_b, outf, N);
        k_seed<<<CDIV(N * 64, B), B, 0, stream>>>(outf, h, outb, N * 64);
    }

    // edge hidden layer (once: independent of steps)
    k_edge_h1<<<CDIV(E, B), B, 0, stream>>>(edge_attr, en1_w, en1_b, h1b, E);

    // message-passing steps
    for (int step = 0; step < 2; ++step) {
        k_zero<<<CDIV(N * 64, B), B, 0, stream>>>(agg, N * 64);
        k_msg<<<CDIV(E, 128), B, 0, stream>>>(outb, h1b, src, dst, Tpk, agg, E);
        k_m<<<CDIV(N * 64, B), B, 0, stream>>>(agg, cnt, conv_b, N);
        k_linh<<<CDIV(N * 64, B), B, 0, stream>>>(h, linh_w, linh_b, hnew, N);
        k_linhm<<<CDIV(N * 64, B), B, 0, stream>>>(hnew, agg, h, linhm_w, linhm_b,
                                                   outf, outb, N);
        float* t = h; h = hnew; hnew = t;   // h <- hnew for next step
    }

    k_final<<<CDIV(N * 64, B), B, 0, stream>>>(outf, (float*)d_out, N * 64);
}
